// NoSoftmaxAttention_59133109731524
// MI455X (gfx1250) — matrix operand order repaired and checked by reading, not compiled, not run on hardware
//
#include <hip/hip_runtime.h>
#include <hip/hip_bf16.h>

typedef __attribute__((ext_vector_type(16))) __bf16 v16bf;
typedef __attribute__((ext_vector_type(8)))  __bf16 v8bf;
typedef __attribute__((ext_vector_type(8)))  float  v8f;

// ---------------------------------------------------------------------------
// gfx1250 async global->LDS copy (ASYNCcnt path) via inline asm:
//   GLOBAL_LOAD_ASYNC_TO_LDS_B128  vdst(lds byte addr), vaddr(64b global), off
// Generic LDS pointers carry the LDS offset in addr[31:0] (aperture rules),
// so truncating the generic pointer yields the VDST operand.
// ---------------------------------------------------------------------------
__device__ __forceinline__ void async_copy_b128(const __bf16* g, __bf16* l) {
    unsigned loff = (unsigned)(uintptr_t)l;            // LDS byte offset (addr[31:0])
    unsigned long long ga = (unsigned long long)(uintptr_t)g;
    asm volatile("global_load_async_to_lds_b128 %0, %1, off"
                 :: "v"(loff), "v"(ga) : "memory");
}

__device__ __forceinline__ void wait_async0() {
#if defined(__gfx1250__) && __has_builtin(__builtin_amdgcn_s_wait_asynccnt)
    __builtin_amdgcn_s_wait_asynccnt(0);
#else
    asm volatile("s_wait_asynccnt 0x0" ::: "memory");
#endif
}

// ---------------------------------------------------------------------------
// f32 -> bf16 bulk convert (8 elems / thread, 16B stores)
// ---------------------------------------------------------------------------
__global__ __launch_bounds__(256) void cvt_f32_bf16(const float* __restrict__ in,
                                                    __bf16* __restrict__ out,
                                                    long n) {
    long i = (((long)blockIdx.x * blockDim.x) + threadIdx.x) * 8;
    if (i >= n) return;
    const float4* p = (const float4*)(in + i);
    float4 x0 = p[0];
    float4 x1 = p[1];
    v8bf o;
    o[0] = (__bf16)x0.x; o[1] = (__bf16)x0.y; o[2] = (__bf16)x0.z; o[3] = (__bf16)x0.w;
    o[4] = (__bf16)x1.x; o[5] = (__bf16)x1.y; o[6] = (__bf16)x1.z; o[7] = (__bf16)x1.w;
    *(v8bf*)(out + i) = o;
}

// ---------------------------------------------------------------------------
// Generic batched NT GEMM:  C[m,n] = alpha * sum_k A[m,k]*B[n,k]  (+ bias[n])
// Block tile 128x128, BLK_K=32, 8 waves (2x4), wave tile 64x32 -> 8 WMMAs/step.
// Double-buffered LDS fed by async global->LDS copies (ASYNCcnt).
// STORE: 0 = bf16 row-major, 1 = bf16 transposed (C[n*ldc + m]), 2 = f32.
// ---------------------------------------------------------------------------
template <int STORE>
__global__ __launch_bounds__(256) void gemm_nt(
    const __bf16* __restrict__ A, int lda, long strideA,
    const __bf16* __restrict__ B, int ldb, long strideB,
    const float* __restrict__ bias,
    void* __restrict__ Cvoid, int ldc, long strideC,
    int K, float alpha) {

    constexpr int LDS_STRIDE = 40;            // bf16 elems: 80B rows, 16B aligned
    constexpr int BUF = 128 * LDS_STRIDE;     // elems per buffer
    __shared__ __bf16 sA[2 * BUF];
    __shared__ __bf16 sB[2 * BUF];

    const int t    = threadIdx.x;
    const int lane = t & 31;
    const int w    = t >> 5;
    const int waveM = w >> 2;                 // 0..1
    const int waveN = w & 3;                  // 0..3
    const int l16  = lane & 15;
    const int half = lane >> 4;               // 0 | 1

    const long zb = blockIdx.z;
    A += zb * strideA;
    B += zb * strideB;
    const int bM = blockIdx.y * 128;
    const int bN = blockIdx.x * 128;

    // tile-load mapping: 512 16B-chunks per operand, 2 per thread
    const int ldRow = t >> 2;                 // 0..63
    const int ldCol = (t & 3) * 8;            // bf16 K-offset
    const __bf16* gA = A + (size_t)(bM + ldRow) * lda + ldCol;
    const __bf16* gB = B + (size_t)(bN + ldRow) * ldb + ldCol;
    const int sRow = ldRow * LDS_STRIDE + ldCol;

    auto issue_tile = [&](int k0, int buf) {
        __bf16* dA = &sA[buf * BUF + sRow];
        __bf16* dB = &sB[buf * BUF + sRow];
        async_copy_b128(gA + k0, dA);
        async_copy_b128(gA + (size_t)64 * lda + k0, dA + 64 * LDS_STRIDE);
        async_copy_b128(gB + k0, dB);
        async_copy_b128(gB + (size_t)64 * ldb + k0, dB + 64 * LDS_STRIDE);
    };

    v8f acc[4][2];
#pragma unroll
    for (int mi = 0; mi < 4; ++mi)
#pragma unroll
        for (int ni = 0; ni < 2; ++ni) acc[mi][ni] = (v8f){};

    // prologue: land tile 0 in buffer 0
    issue_tile(0, 0);
    wait_async0();
    __syncthreads();

    int buf = 0;
    for (int k0 = 0; k0 < K; k0 += 32) {
        if (k0 + 32 < K) issue_tile(k0 + 32, buf ^ 1);   // overlap with WMMA below

        // A fragments: ISA 16-bit A layout — lane half picks K 0..7/16..23 vs 8..15/24..31
        v16bf aReg[4];
#pragma unroll
        for (int mi = 0; mi < 4; ++mi) {
            const __bf16* p =
                &sA[buf * BUF + (waveM * 64 + mi * 16 + l16) * LDS_STRIDE + half * 8];
            v8bf lo = *(const v8bf*)p;
            v8bf hi = *(const v8bf*)(p + 16);
            aReg[mi] = __builtin_shufflevector(lo, hi, 0, 1, 2, 3, 4, 5, 6, 7,
                                               8, 9, 10, 11, 12, 13, 14, 15);
        }
        // B fragments: lane half picks K 0..7/16..23 vs 8..15/24..31 (matches A)
        v16bf bReg[2];
#pragma unroll
        for (int ni = 0; ni < 2; ++ni) {
            const __bf16* p =
                &sB[buf * BUF + (waveN * 32 + ni * 16 + l16) * LDS_STRIDE + half * 8];
            v8bf lo = *(const v8bf*)p;
            v8bf hi = *(const v8bf*)(p + 16);
            bReg[ni] = __builtin_shufflevector(lo, hi, 0, 1, 2, 3, 4, 5, 6, 7,
                                               8, 9, 10, 11, 12, 13, 14, 15);
        }

#pragma unroll
        for (int mi = 0; mi < 4; ++mi)
#pragma unroll
            for (int ni = 0; ni < 2; ++ni)
                acc[mi][ni] = __builtin_amdgcn_wmma_f32_16x16x32_bf16(
                    false, aReg[mi], false, bReg[ni], (short)0, acc[mi][ni],
                    false, false);

        wait_async0();        // next tile landed in LDS (this wave's copies)
        __syncthreads();      // ... and everyone else's; buffer swap is safe
        buf ^= 1;
    }

    // ------------------------- epilogue -------------------------
    const int half8 = half * 8;
#pragma unroll
    for (int mi = 0; mi < 4; ++mi) {
#pragma unroll
        for (int ni = 0; ni < 2; ++ni) {
            const int n = bN + waveN * 32 + ni * 16 + l16;
            const float bv = bias ? bias[n] : 0.0f;
            const int mBase = bM + waveM * 64 + mi * 16 + half8;
            if constexpr (STORE == 2) {
                float* Cf = (float*)Cvoid + zb * strideC;
#pragma unroll
                for (int j = 0; j < 8; ++j)
                    Cf[(size_t)(mBase + j) * ldc + n] = acc[mi][ni][j] * alpha + bv;
            } else if constexpr (STORE == 0) {
                __bf16* Cb = (__bf16*)Cvoid + zb * strideC;
#pragma unroll
                for (int j = 0; j < 8; ++j)
                    Cb[(size_t)(mBase + j) * ldc + n] =
                        (__bf16)(acc[mi][ni][j] * alpha + bv);
            } else {  // transposed bf16 store: one contiguous b128 per lane
                __bf16* Cb = (__bf16*)Cvoid + zb * strideC;
                v8bf o;
#pragma unroll
                for (int j = 0; j < 8; ++j)
                    o[j] = (__bf16)(acc[mi][ni][j] * alpha + bv);
                *(v8bf*)&Cb[(size_t)n * ldc + mBase] = o;
            }
        }
    }
}

// ---------------------------------------------------------------------------
// Host: x,Wq,bq,Wk,bk,Wv,bv,Wo,bo  ->  y  (B=4, S=2048, E=1024)
// ---------------------------------------------------------------------------
extern "C" void kernel_launch(void* const* d_in, const int* in_sizes, int n_in,
                              void* d_out, int out_size, void* d_ws, size_t ws_size,
                              hipStream_t stream) {
    (void)in_sizes; (void)n_in; (void)out_size; (void)ws_size;
    const float* x  = (const float*)d_in[0];
    const float* Wq = (const float*)d_in[1];
    const float* bq = (const float*)d_in[2];
    const float* Wk = (const float*)d_in[3];
    const float* bk = (const float*)d_in[4];
    const float* Wv = (const float*)d_in[5];
    const float* bv = (const float*)d_in[6];
    const float* Wo = (const float*)d_in[7];
    const float* bo = (const float*)d_in[8];
    float* y = (float*)d_out;

    const long S = 2048, E = 1024, M = 4 * S;  // 8192
    const size_t MB = 1u << 20;
    char* w = (char*)d_ws;
    __bf16* xb  = (__bf16*)(w + 0 * MB);     // 16 MB  [8192,1024]
    __bf16* Wqb = (__bf16*)(w + 16 * MB);    //  2 MB
    __bf16* Wkb = (__bf16*)(w + 18 * MB);
    __bf16* Wvb = (__bf16*)(w + 20 * MB);
    __bf16* Wob = (__bf16*)(w + 22 * MB);
    __bf16* Qb  = (__bf16*)(w + 24 * MB);    // 16 MB  [8192,1024]
    __bf16* Kb  = (__bf16*)(w + 40 * MB);    // 16 MB
    __bf16* Vtb = (__bf16*)(w + 56 * MB);    // 16 MB  [4][1024][2048]  (V^T per batch)
    __bf16* Sb  = (__bf16*)(w + 72 * MB);    // 32 MB  [4][2048][2048]  scaled scores
    __bf16* Ob  = (__bf16*)(w + 104 * MB);   // 16 MB  [8192,1024]      attn out

    dim3 blk(256);
    cvt_f32_bf16<<<(M * E) / 2048, blk, 0, stream>>>(x, xb, M * E);
    cvt_f32_bf16<<<(E * E) / 2048, blk, 0, stream>>>(Wq, Wqb, E * E);
    cvt_f32_bf16<<<(E * E) / 2048, blk, 0, stream>>>(Wk, Wkb, E * E);
    cvt_f32_bf16<<<(E * E) / 2048, blk, 0, stream>>>(Wv, Wvb, E * E);
    cvt_f32_bf16<<<(E * E) / 2048, blk, 0, stream>>>(Wo, Wob, E * E);

    // Q = x Wq^T + bq ; K = x Wk^T + bk       (M=8192, N=1024, K=1024)
    gemm_nt<0><<<dim3(8, 64, 1), blk, 0, stream>>>(xb, 1024, 0, Wqb, 1024, 0, bq,
                                                   Qb, 1024, 0, 1024, 1.0f);
    gemm_nt<0><<<dim3(8, 64, 1), blk, 0, stream>>>(xb, 1024, 0, Wkb, 1024, 0, bk,
                                                   Kb, 1024, 0, 1024, 1.0f);
    // V^T[e][s] = (x Wv^T + bv)^T  per batch  (M=2048/batch, N=1024, K=1024)
    gemm_nt<1><<<dim3(8, 16, 4), blk, 0, stream>>>(xb, 1024, 2048L * 1024,
                                                   Wvb, 1024, 0, bv,
                                                   Vtb, 2048, 1024L * 2048, 1024, 1.0f);
    // scores = (Q K^T) * 0.125   per batch    (M=2048, N=2048, K=1024)
    gemm_nt<0><<<dim3(16, 16, 4), blk, 0, stream>>>(Qb, 1024, 2048L * 1024,
                                                    Kb, 1024, 2048L * 1024, nullptr,
                                                    Sb, 2048, 2048L * 2048, 1024, 0.125f);
    // out = scores V = scores (V^T)^T  per batch  (M=2048, N=1024, K=2048)
    gemm_nt<0><<<dim3(8, 16, 4), blk, 0, stream>>>(Sb, 2048, 2048L * 2048,
                                                   Vtb, 2048, 1024L * 2048, nullptr,
                                                   Ob, 1024, 2048L * 1024, 2048, 1.0f);
    // y = out Wo^T + bo  (f32)                (M=8192, N=1024, K=1024)
    gemm_nt<2><<<dim3(8, 64, 1), blk, 0, stream>>>(Ob, 1024, 0, Wob, 1024, 0, bo,
                                                   y, 1024, 0, 1024, 1.0f);
}
